// NeuralEmbeddingLayer_65498251264496
// MI455X (gfx1250) — compile-verified, matrix operand order checked by reading, not run on hardware
//
#include <hip/hip_runtime.h>
#include <cmath>

// ---------------- problem constants (from reference) ----------------
#define B_     32
#define T_     512
#define C_     384
#define D_     1536        // C*MULT
#define H_     1024
#define ND_    16
#define L_     255         // (T-SIZE)/STRIDE + 1
#define KD_    6144        // SIZE*D
#define SCALE_ 32.0f       // sqrt(H)

typedef __attribute__((ext_vector_type(16))) __bf16 v16bf;
typedef __attribute__((ext_vector_type(8)))  __bf16 bf8;
typedef __attribute__((ext_vector_type(8)))  float  v8f;

// ---------------------------------------------------------------------------
// Prep: elementwise f32 -> bf16 (hardware cvt, 8 elems/thread, b128 in/out)
// ---------------------------------------------------------------------------
__global__ void cvt_f32_bf16_kernel(const float* __restrict__ src,
                                    __bf16* __restrict__ dst, int n8) {
    const int i = blockIdx.x * blockDim.x + threadIdx.x;
    if (i >= n8) return;
    const float4* p = reinterpret_cast<const float4*>(src) + 2 * (size_t)i;
    float4 f0 = p[0], f1 = p[1];
    bf8 o;
    o[0] = (__bf16)f0.x; o[1] = (__bf16)f0.y; o[2] = (__bf16)f0.z; o[3] = (__bf16)f0.w;
    o[4] = (__bf16)f1.x; o[5] = (__bf16)f1.y; o[6] = (__bf16)f1.z; o[7] = (__bf16)f1.w;
    reinterpret_cast<bf8*>(dst)[(size_t)i] = o;
}

// WMMA 16-bit operand fragment from a row-major bf16 matrix (ISA 7.12.2):
// lane's row = lane&15; lanes 0-15 hold K=[k..k+7],[k+16..k+23]; lanes 16-31
// hold K=[k+8..k+15],[k+24..k+31]. Two b128 loads + register concat.
__device__ __forceinline__ v16bf load_frag(const __bf16* __restrict__ rowbase,
                                           int k, int hi) {
    bf8 c0 = *reinterpret_cast<const bf8*>(rowbase + k + hi * 8);
    bf8 c1 = *reinterpret_cast<const bf8*>(rowbase + k + 16 + hi * 8);
    return __builtin_shufflevector(c0, c1, 0, 1, 2, 3, 4, 5, 6, 7,
                                   8, 9, 10, 11, 12, 13, 14, 15);
}

__device__ __forceinline__ v8f wmma_bf16(v16bf a, v16bf b, v8f c) {
    return __builtin_amdgcn_wmma_f32_16x16x32_bf16(
        false, a, false, b, (short)0, c, false, false);
}

__device__ __forceinline__ float gelu_scale(float v) {
    return 0.5f * v * (1.0f + erff(v * 0.70710678118654752f)) * SCALE_;
}

// ---------------------------------------------------------------------------
// Kernel 1: x[b,t,d] = gelu(spikes[b]@W_embed[date[b]]^T + b_embed) * 32
// Block = 128 threads (4 waves). Block tile: 32 rows(t) x 128 cols(d); each
// wave owns a 32x32 register tile (2M x 2N frags -> 4 WMMA per 8 b128 loads).
// Output stored as bf16 (feeds kernel 2's A operand directly).
// ---------------------------------------------------------------------------
__global__ void embed_gelu_kernel(const __bf16* __restrict__ spikes_bf,
                                  const int*    __restrict__ date_idx,
                                  const __bf16* __restrict__ Wemb_bf,
                                  const float*  __restrict__ b_embed,
                                  __bf16*       __restrict__ xbuf) {
    const int wave = threadIdx.x >> 5;
    const int lane = threadIdx.x & 31;
    const int hi   = lane >> 4;
    const int lo   = lane & 15;

    const int b  = blockIdx.z;
    const int t0 = blockIdx.y * 32;                // two M tiles: t0, t0+16
    const int d0 = blockIdx.x * 128 + wave * 32;   // two N tiles: d0, d0+16

    const int di = date_idx[b];
    const __bf16* __restrict__ A0 = spikes_bf + ((size_t)b * T_ + (t0 + lo)) * C_;
    const __bf16* __restrict__ A1 = A0 + (size_t)16 * C_;
    const __bf16* __restrict__ B0 = Wemb_bf + ((size_t)di * D_ + (d0 + lo)) * C_;
    const __bf16* __restrict__ B1 = B0 + (size_t)16 * C_;

    v8f acc00 = {}, acc01 = {}, acc10 = {}, acc11 = {};
#pragma unroll 2
    for (int k = 0; k < C_; k += 32) {
        v16bf a0 = load_frag(A0, k, hi);
        v16bf a1 = load_frag(A1, k, hi);
        v16bf w0 = load_frag(B0, k, hi);
        v16bf w1 = load_frag(B1, k, hi);
        acc00 = wmma_bf16(a0, w0, acc00);
        acc01 = wmma_bf16(a0, w1, acc01);
        acc10 = wmma_bf16(a1, w0, acc10);
        acc11 = wmma_bf16(a1, w1, acc11);
    }

    // Epilogue: C/D layout -> lane<16: M=r, lane>=16: M=8+r; N = lane&15.
    const float bias0 = b_embed[(size_t)di * D_ + d0 + lo];
    const float bias1 = b_embed[(size_t)di * D_ + d0 + 16 + lo];
    __bf16* __restrict__ o0 =
        xbuf + ((size_t)b * T_ + (t0 + hi * 8)) * D_ + d0 + lo;
    __bf16* __restrict__ o1 = o0 + (size_t)16 * D_;
#pragma unroll
    for (int r = 0; r < 8; ++r) {
        o0[(size_t)r * D_]      = (__bf16)gelu_scale(acc00[r] + bias0);
        o0[(size_t)r * D_ + 16] = (__bf16)gelu_scale(acc01[r] + bias1);
        o1[(size_t)r * D_]      = (__bf16)gelu_scale(acc10[r] + bias0);
        o1[(size_t)r * D_ + 16] = (__bf16)gelu_scale(acc11[r] + bias1);
    }
}

// ---------------------------------------------------------------------------
// Kernel 2: out[b,l,h] = sum_{s,d} x[b,2l+s,d]*W_proj[h,s*D+d] + b_proj[h]
//                        + pos_table[ts[b,l],h]
// Block = 128 threads (4 waves). Block tile: 32 rows(l) x 128 cols(h); each
// wave owns a 32x32 register tile. K = 6144 (chunks of 32 never straddle a
// window element since D % 32 == 0). A operand = bf16 xbuf from kernel 1.
// ---------------------------------------------------------------------------
__global__ void proj_pos_kernel(const __bf16* __restrict__ xbuf,
                                const __bf16* __restrict__ Wproj_bf,
                                const float*  __restrict__ b_proj,
                                const float*  __restrict__ pos_table,
                                const int*    __restrict__ ts_in,
                                float*        __restrict__ out_x) {
    const int wave = threadIdx.x >> 5;
    const int lane = threadIdx.x & 31;
    const int hi   = lane >> 4;
    const int lo   = lane & 15;

    const int b  = blockIdx.z;
    const int l0 = blockIdx.y * 32;                // two M tiles: l0, l0+16
    const int h0 = blockIdx.x * 128 + wave * 32;   // two N tiles: h0, h0+16

    int lr0 = l0 + lo;       if (lr0 >= L_) lr0 = L_ - 1;  // clamp ragged edge
    int lr1 = l0 + 16 + lo;  if (lr1 >= L_) lr1 = L_ - 1;

    const __bf16* __restrict__ B0 = Wproj_bf + (size_t)(h0 + lo) * KD_;
    const __bf16* __restrict__ B1 = B0 + (size_t)16 * KD_;

    v8f acc00 = {}, acc01 = {}, acc10 = {}, acc11 = {};
#pragma unroll
    for (int s = 0; s < 4; ++s) {
        const __bf16* __restrict__ A0 =
            xbuf + ((size_t)b * T_ + (2 * lr0 + s)) * D_;
        const __bf16* __restrict__ A1 =
            xbuf + ((size_t)b * T_ + (2 * lr1 + s)) * D_;
#pragma unroll 2
        for (int dj = 0; dj < D_; dj += 32) {
            v16bf a0 = load_frag(A0, dj, hi);
            v16bf a1 = load_frag(A1, dj, hi);
            v16bf w0 = load_frag(B0, s * D_ + dj, hi);
            v16bf w1 = load_frag(B1, s * D_ + dj, hi);
            acc00 = wmma_bf16(a0, w0, acc00);
            acc01 = wmma_bf16(a0, w1, acc01);
            acc10 = wmma_bf16(a1, w0, acc10);
            acc11 = wmma_bf16(a1, w1, acc11);
        }
    }

    const float bias0 = b_proj[h0 + lo];
    const float bias1 = b_proj[h0 + 16 + lo];
#pragma unroll
    for (int mt = 0; mt < 2; ++mt) {
        const v8f* accA = mt ? &acc10 : &acc00;
        const v8f* accB = mt ? &acc11 : &acc01;
#pragma unroll
        for (int r = 0; r < 8; ++r) {
            const int l = l0 + mt * 16 + r + hi * 8;
            if (l < L_) {
                const int    tval = ts_in[(size_t)b * T_ + l];
                const float* pt   = pos_table + (size_t)tval * H_;
                float* o = out_x + ((size_t)b * L_ + l) * H_ + h0 + lo;
                o[0]  = (*accA)[r] + bias0 + pt[h0 + lo];
                o[16] = (*accB)[r] + bias1 + pt[h0 + 16 + lo];
            }
        }
    }
}

// ---------------------------------------------------------------------------
// Kernel 3: mask[b,l] = prod_{s<4} spikes_mask[b,2l+s];  ts[b,l] copy.
// ---------------------------------------------------------------------------
__global__ void mask_ts_kernel(const int* __restrict__ mask_in,
                               const int* __restrict__ ts_in,
                               float* __restrict__ out_mask,
                               float* __restrict__ out_ts) {
    const int idx = blockIdx.x * blockDim.x + threadIdx.x;
    if (idx >= B_ * L_) return;
    const int b = idx / L_;
    const int l = idx - b * L_;
    const int base = b * T_ + 2 * l;
    const int m = mask_in[base] * mask_in[base + 1] *
                  mask_in[base + 2] * mask_in[base + 3];
    out_mask[idx] = (float)m;
    out_ts[idx]   = (float)ts_in[b * T_ + l];
}

// ---------------------------------------------------------------------------
extern "C" void kernel_launch(void* const* d_in, const int* in_sizes, int n_in,
                              void* d_out, int out_size, void* d_ws, size_t ws_size,
                              hipStream_t stream) {
    const float* spikes     = (const float*)d_in[0];
    const int*   spikes_msk = (const int*)  d_in[1];
    const int*   spikes_ts  = (const int*)  d_in[2];
    const int*   date_idx   = (const int*)  d_in[3];
    const float* W_embed    = (const float*)d_in[4];
    const float* b_embed    = (const float*)d_in[5];
    const float* W_proj     = (const float*)d_in[6];
    const float* b_proj     = (const float*)d_in[7];
    const float* pos_table  = (const float*)d_in[8];

    float* out = (float*)d_out;
    const size_t XN = (size_t)B_ * L_ * H_;   // 8,355,840
    float* out_mask = out + XN;
    float* out_ts   = out + XN + (size_t)B_ * L_;

    // --- workspace layout (bf16), ~94 MB total ---
    const size_t N_SPK  = (size_t)B_ * T_ * C_;   //  6,291,456
    const size_t N_WEMB = (size_t)ND_ * D_ * C_;  //  9,437,184
    const size_t N_WPRJ = (size_t)H_ * KD_;       //  6,291,456
    __bf16* spikes_bf = (__bf16*)d_ws;
    __bf16* Wemb_bf   = spikes_bf + N_SPK;
    __bf16* Wproj_bf  = Wemb_bf + N_WEMB;
    __bf16* xbuf      = Wproj_bf + N_WPRJ;        // B*T*D bf16

    // Stage 0: one-time precision downconversions (each value reused 4-512x)
    {
        const int t = 256;
        cvt_f32_bf16_kernel<<<(int)((N_SPK / 8 + t - 1) / t), t, 0, stream>>>(
            spikes, spikes_bf, (int)(N_SPK / 8));
        cvt_f32_bf16_kernel<<<(int)((N_WEMB / 8 + t - 1) / t), t, 0, stream>>>(
            W_embed, Wemb_bf, (int)(N_WEMB / 8));
        cvt_f32_bf16_kernel<<<(int)((N_WPRJ / 8 + t - 1) / t), t, 0, stream>>>(
            W_proj, Wproj_bf, (int)(N_WPRJ / 8));
    }

    // Stage 1: per-batch embed GEMM + gelu*scale (bf16 out)
    dim3 g1(D_ / 128, T_ / 32, B_);           // (12, 16, 32)
    embed_gelu_kernel<<<g1, dim3(128), 0, stream>>>(
        spikes_bf, date_idx, Wemb_bf, b_embed, xbuf);

    // Stage 2: windowed projection GEMM + bias + pos_table gather
    dim3 g2(H_ / 128, (L_ + 31) / 32, B_);    // (8, 8, 32)
    proj_pos_kernel<<<g2, dim3(128), 0, stream>>>(
        xbuf, Wproj_bf, b_proj, pos_table, spikes_ts, out);

    // Stage 3: mask product + ts copy
    const int n3 = B_ * L_;
    mask_ts_kernel<<<(n3 + 255) / 256, 256, 0, stream>>>(
        spikes_msk, spikes_ts, out_mask, out_ts);
}
